// Attention_5609227288673
// MI455X (gfx1250) — matrix operand order repaired and checked by reading, not compiled, not run on hardware
//
#include <hip/hip_runtime.h>
#include <hip/hip_fp16.h>

// MHA forward for MI455X (gfx1250, wave32).
// Pipeline: f32->f16 pre-convert (weights once, activations per projection)
// -> pure-f16 WMMA QKV projection (V stored transposed) -> flash attention
// (WMMA, online softmax, LDS transpose of P) -> WMMA output projection.

typedef __attribute__((ext_vector_type(16))) _Float16 v16h;
typedef __attribute__((ext_vector_type(8)))  _Float16 v8h;
typedef __attribute__((ext_vector_type(8)))  float    v8f;

#define DM   1024   // d_model
#define NH   16     // heads
#define DK   64     // head dim
#define NQ   2048   // seq len (q and k)
#define NBAT 2      // batch
#define MROWS (NBAT * NQ)   // 4096 flattened rows

// ---- WMMA fragment loaders (layouts per CDNA5 ISA 7.12.2) ----------------

// A-matrix 16x32 f16 from row-major f16 source.
// lane<16:  m=lane,    K = kb+[0..7], kb+[16..23]
// lane>=16: m=lane-16, K = kb+[8..15], kb+[24..31]
__device__ __forceinline__ v16h ldA_f16(const _Float16* __restrict__ base, int lda, int kb) {
  const int lane = threadIdx.x & 31;
  const int m    = lane & 15;
  const int koff = (lane >> 4) << 3;
  const _Float16* r = base + (size_t)m * lda + kb + koff;
  v16h a;
#pragma unroll
  for (int i = 0; i < 8; ++i) { a[i] = r[i]; a[i + 8] = r[i + 16]; }
  return a;
}

// B-matrix 32x16 f16 where column n is contiguous in memory:
// B[k][n] = base[n*ldb + k]
// lane<16: n=lane, K=kb+[0..7], kb+[16..23]; lane>=16: n=lane-16, K=kb+[8..15], kb+[24..31]
__device__ __forceinline__ v16h ldB_rowk_f16(const _Float16* __restrict__ base, int ldb, int kb) {
  const int lane = threadIdx.x & 31;
  const int n    = lane & 15;
  const int koff = (lane >> 4) << 3;
  const _Float16* r = base + (size_t)n * ldb + kb + koff;
  v16h b;
#pragma unroll
  for (int i = 0; i < 8; ++i) { b[i] = r[i]; b[i + 8] = r[i + 16]; }
  return b;
}

__device__ __forceinline__ v8f wmma_f16(v16h a, v16h b, v8f c) {
  return __builtin_amdgcn_wmma_f32_16x16x32_f16(false, a, false, b, (short)0, c,
                                                false, false);
}

// ---- Kernel 0: elementwise fp32 -> f16 (n must be a multiple of 2048) ----
__global__ __launch_bounds__(256) void cvt_f16_kernel(
    const float* __restrict__ src, _Float16* __restrict__ dst) {
  const size_t i = ((size_t)blockIdx.x * 256 + threadIdx.x) * 8;
  const float4 a = *(const float4*)(src + i);
  const float4 b = *(const float4*)(src + i + 4);
  v8h h;
  h[0] = (_Float16)a.x; h[1] = (_Float16)a.y;
  h[2] = (_Float16)a.z; h[3] = (_Float16)a.w;
  h[4] = (_Float16)b.x; h[5] = (_Float16)b.y;
  h[6] = (_Float16)b.z; h[7] = (_Float16)b.w;
  *(v8h*)(dst + i) = h;
}

// ---- Kernel 1: QKV projection  out = X @ W^T + bias  (pure f16 GEMM) -----
// One wave computes a 16(M) x 64(N) tile; N-tiles align exactly with heads.
// vtrans==0: out16[(b*NH+h)*NQ + n][d]   (Q, K layout)
// vtrans==1: out16[(b*NH+h)*DK + d][n]   (V^T layout for coalesced P@V reads)
__global__ __launch_bounds__(128) void qkv_proj_kernel(
    const _Float16* __restrict__ X16,  // (MROWS, DM) f16
    const _Float16* __restrict__ W16,  // (DM, DM) f16 row-major (out, in)
    const float* __restrict__ bias,    // (DM) fp32
    _Float16* __restrict__ out16,
    int vtrans)
{
  const int wave  = (int)((blockIdx.x * blockDim.x + threadIdx.x) >> 5);
  const int mtile = wave >> 4;       // 0..255
  const int ntile = wave & 15;       // 0..15  (== head index)
  const int lane  = threadIdx.x & 31;
  const int row0  = mtile * 16;

  v8f acc[4] = {};
  const _Float16* A0 = X16 + (size_t)row0 * DM;

  for (int kb = 0; kb < DM; kb += 32) {
    if (kb + 32 < DM) __builtin_prefetch(A0 + (size_t)(lane & 15) * DM + kb + 32, 0, 1);
    v16h a = ldA_f16(A0, DM, kb);
#pragma unroll
    for (int j = 0; j < 4; ++j) {
      v16h b = ldB_rowk_f16(W16 + (size_t)(ntile * 64 + j * 16) * DM, DM, kb);
      acc[j] = wmma_f16(a, b, acc[j]);
    }
  }

  const int n  = lane & 15;
  const int mb = (lane >> 4) * 8;
  const int h  = ntile;
#pragma unroll
  for (int j = 0; j < 4; ++j) {
    const int d  = j * 16 + n;                // dim within head
    const float bf = bias[ntile * 64 + d];
#pragma unroll
    for (int r = 0; r < 8; ++r) {
      const int g  = row0 + mb + r;           // flattened (b, n) row
      const int bb = g >> 11;                 // / NQ
      const int nn = g & (NQ - 1);
      const _Float16 val = (_Float16)(acc[j][r] + bf);
      if (!vtrans)
        out16[(((size_t)(bb * NH + h)) * NQ + nn) * DK + d] = val;
      else
        out16[(((size_t)(bb * NH + h)) * DK + d) * NQ + nn] = val;
    }
  }
}

// ---- Kernel 2: flash attention per (b, h, 16-row q tile) -----------------
__global__ __launch_bounds__(128) void attn_kernel(
    const _Float16* __restrict__ Q16,  // (NBAT, NH, NQ, DK)
    const _Float16* __restrict__ K16,  // (NBAT, NH, NQ, DK)
    const _Float16* __restrict__ Vt16, // (NBAT, NH, DK, NQ)  V transposed
    _Float16* __restrict__ Ctx16)      // (NBAT, NQ, DM) row-major context
{
  __shared__ _Float16 plds[4][16 * 32];    // per-wave P transpose staging
  const int widb = threadIdx.x >> 5;
  const int wave = (int)(blockIdx.x * 4 + widb);
  const int qt   = wave & 127;             // q tile index (0..127)
  const int bh   = wave >> 7;              // 0..31 == b*NH + h
  const int h    = bh & (NH - 1);
  const int b    = bh >> 4;
  const int lane = threadIdx.x & 31;
  const int n    = lane & 15;
  const int hlf  = lane >> 4;              // 0/1

  const _Float16* Qb = Q16  + ((size_t)bh * NQ + (size_t)qt * 16) * DK;
  const _Float16* Kb = K16  + (size_t)bh * NQ * DK;
  const _Float16* Vt = Vt16 + (size_t)bh * DK * NQ;

  // Q fragments (d-tiles 0..31 and 32..63), kept in registers
  v16h qa0 = ldA_f16(Qb, DK, 0);
  v16h qa1 = ldA_f16(Qb, DK, 32);

  v8f o[4] = {};
  float mrow[8], lrow[8];
#pragma unroll
  for (int r = 0; r < 8; ++r) { mrow[r] = -1e30f; lrow[r] = 0.0f; }

  _Float16* pl = plds[widb];
  const float scale = 0.125f;              // 1/sqrt(64)

  for (int kb = 0; kb < NQ; kb += 32) {
    if (kb + 32 < NQ)
      __builtin_prefetch(Kb + (size_t)(kb + 32 + n) * DK, 0, 1);

    // ---- S = (Q K^T) tile: 16 x 32, as two 16x16 accumulators
    v8f s0 = {}, s1 = {};
    s0 = wmma_f16(qa0, ldB_rowk_f16(Kb + (size_t)kb * DK,        DK, 0),  s0);
    s0 = wmma_f16(qa1, ldB_rowk_f16(Kb + (size_t)kb * DK,        DK, 32), s0);
    s1 = wmma_f16(qa0, ldB_rowk_f16(Kb + (size_t)(kb + 16) * DK, DK, 0),  s1);
    s1 = wmma_f16(qa1, ldB_rowk_f16(Kb + (size_t)(kb + 16) * DK, DK, 32), s1);

    // ---- online softmax update (row = r + 8*hlf, cols across 16 lanes)
    float corr[8];
#pragma unroll
    for (int r = 0; r < 8; ++r) {
      float a0 = s0[r] * scale, a1 = s1[r] * scale;
      float t = fmaxf(a0, a1);
      t = fmaxf(t, __shfl_xor(t, 1, 32));
      t = fmaxf(t, __shfl_xor(t, 2, 32));
      t = fmaxf(t, __shfl_xor(t, 4, 32));
      t = fmaxf(t, __shfl_xor(t, 8, 32));
      const float mnew = fmaxf(mrow[r], t);
      corr[r] = __expf(mrow[r] - mnew);
      mrow[r] = mnew;
      const float p0 = __expf(a0 - mnew);
      const float p1 = __expf(a1 - mnew);
      float sum = p0 + p1;
      sum += __shfl_xor(sum, 1, 32);
      sum += __shfl_xor(sum, 2, 32);
      sum += __shfl_xor(sum, 4, 32);
      sum += __shfl_xor(sum, 8, 32);
      lrow[r] = lrow[r] * corr[r] + sum;
      s0[r] = p0; s1[r] = p1;
    }
#pragma unroll
    for (int j = 0; j < 4; ++j)
#pragma unroll
      for (int r = 0; r < 8; ++r) o[j][r] *= corr[r];

    // ---- transpose P (C layout -> A layout) through per-wave LDS
#pragma unroll
    for (int r = 0; r < 8; ++r) {
      const int row = r + 8 * hlf;
      pl[row * 32 + n]      = (_Float16)s0[r];
      pl[row * 32 + 16 + n] = (_Float16)s1[r];
    }
    __asm volatile("s_wait_dscnt 0x0" ::: "memory");
    v16h pa;
    {
      const int koff = hlf * 8;
#pragma unroll
      for (int i = 0; i < 8; ++i) {
        pa[i]     = pl[n * 32 + koff + i];
        pa[i + 8] = pl[n * 32 + 16 + koff + i];
      }
    }
    __asm volatile("s_wait_dscnt 0x0" ::: "memory");

    // ---- O += P @ V: V^T gives contiguous-k B columns (b128 loads)
#pragma unroll
    for (int j = 0; j < 4; ++j) {
      v16h vb = ldB_rowk_f16(Vt + (size_t)(j * 16) * NQ, NQ, kb);
      o[j] = wmma_f16(pa, vb, o[j]);
    }
  }

  // ---- normalize and store context (b, q, h*64 + d) f16
#pragma unroll
  for (int j = 0; j < 4; ++j) {
#pragma unroll
    for (int r = 0; r < 8; ++r) {
      const int qrow = qt * 16 + 8 * hlf + r;
      const float val = o[j][r] / lrow[r];
      Ctx16[((size_t)(b * NQ + qrow)) * DM + h * 64 + j * 16 + n] = (_Float16)val;
    }
  }
}

// ---- Kernel 3: output projection  out = Ctx @ Wo^T + bo  (fp32 out) ------
__global__ __launch_bounds__(128) void out_proj_kernel(
    const _Float16* __restrict__ Ctx16, // (MROWS, DM) f16
    const _Float16* __restrict__ Wo16,  // (DM, DM) f16
    const float* __restrict__ bo,       // (DM)
    float* __restrict__ out)            // (MROWS, DM) fp32
{
  const int wave  = (int)((blockIdx.x * blockDim.x + threadIdx.x) >> 5);
  const int mtile = wave >> 4;
  const int ntile = wave & 15;
  const int lane  = threadIdx.x & 31;
  const int row0  = mtile * 16;

  v8f acc[4] = {};
  const _Float16* A0 = Ctx16 + (size_t)row0 * DM;

  for (int kb = 0; kb < DM; kb += 32) {
    if (kb + 32 < DM) __builtin_prefetch(A0 + (size_t)(lane & 15) * DM + kb + 32, 0, 1);
    v16h a = ldA_f16(A0, DM, kb);
#pragma unroll
    for (int j = 0; j < 4; ++j) {
      v16h b = ldB_rowk_f16(Wo16 + (size_t)(ntile * 64 + j * 16) * DM, DM, kb);
      acc[j] = wmma_f16(a, b, acc[j]);
    }
  }

  const int n  = lane & 15;
  const int mb = (lane >> 4) * 8;
#pragma unroll
  for (int j = 0; j < 4; ++j) {
    const int f = ntile * 64 + j * 16 + n;
    const float bf = bo[f];
#pragma unroll
    for (int r = 0; r < 8; ++r) {
      out[(size_t)(row0 + mb + r) * DM + f] = acc[j][r] + bf;
    }
  }
}

// ---- Host launcher --------------------------------------------------------
extern "C" void kernel_launch(void* const* d_in, const int* in_sizes, int n_in,
                              void* d_out, int out_size, void* d_ws, size_t ws_size,
                              hipStream_t stream) {
  const float* queries = (const float*)d_in[0];
  const float* keys    = (const float*)d_in[1];
  const float* values  = (const float*)d_in[2];
  const float* Wq = (const float*)d_in[3];
  const float* bq = (const float*)d_in[4];
  const float* Wk = (const float*)d_in[5];
  const float* bk = (const float*)d_in[6];
  const float* Wv = (const float*)d_in[7];
  const float* bv = (const float*)d_in[8];
  const float* Wo = (const float*)d_in[9];
  const float* bo = (const float*)d_in[10];
  float* out = (float*)d_out;

  const size_t headElems = (size_t)NBAT * NH * NQ * DK;   // 4,194,304
  const size_t wElems    = (size_t)DM * DM;               // 1,048,576
  _Float16* q16  = (_Float16*)d_ws;
  _Float16* k16  = q16  + headElems;
  _Float16* vt16 = k16  + headElems;        // V^T (b,h,d,n)
  _Float16* c16  = vt16 + headElems;        // context
  _Float16* wq16 = c16  + headElems;
  _Float16* wk16 = wq16 + wElems;
  _Float16* wv16 = wk16 + wElems;
  _Float16* wo16 = wv16 + wElems;
  _Float16* x16  = wo16 + wElems;           // activation staging (reused)
  // total: 4*4M + 4*1M + 4M elems * 2B = 48 MB of ws

  // weights: convert once per launch (L2-resident thereafter)
  cvt_f16_kernel<<<wElems / 2048, 256, 0, stream>>>(Wq, wq16);
  cvt_f16_kernel<<<wElems / 2048, 256, 0, stream>>>(Wk, wk16);
  cvt_f16_kernel<<<wElems / 2048, 256, 0, stream>>>(Wv, wv16);
  cvt_f16_kernel<<<wElems / 2048, 256, 0, stream>>>(Wo, wo16);

  const int actBlocks = (int)(((size_t)MROWS * DM) / 2048);
  // Q projection
  cvt_f16_kernel<<<actBlocks, 256, 0, stream>>>(queries, x16);
  qkv_proj_kernel<<<1024, 128, 0, stream>>>(x16, wq16, bq, q16, 0);
  // K projection
  cvt_f16_kernel<<<actBlocks, 256, 0, stream>>>(keys, x16);
  qkv_proj_kernel<<<1024, 128, 0, stream>>>(x16, wk16, bk, k16, 0);
  // V projection (stored transposed)
  cvt_f16_kernel<<<actBlocks, 256, 0, stream>>>(values, x16);
  qkv_proj_kernel<<<1024, 128, 0, stream>>>(x16, wv16, bv, vt16, 1);

  // flash attention: 2*16*128 = 4096 (b,h,qtile) waves -> 1024 blocks
  attn_kernel<<<1024, 128, 0, stream>>>(q16, k16, vt16, c16);

  // output projection
  out_proj_kernel<<<1024, 128, 0, stream>>>(c16, wo16, bo, out);
}
